// CNN_50740743635248
// MI455X (gfx1250) — compile-verified, matrix operand order checked by reading, not run on hardware
//
#include <hip/hip_runtime.h>
#include <hip/hip_bf16.h>
#include <math.h>

typedef __attribute__((ext_vector_type(16))) _Float16 v16h;
typedef __attribute__((ext_vector_type(8)))  float    v8f;

#define BB   128
#define NN   22
#define CC   200
#define HH   512
#define WW   3
#define LT   25
#define LB   100
#define LOT  23      // title conv output length
#define LOB  98      // body conv output length
#define KTOT 600     // C*W
#define KT   19      // ceil(608/32) K-tiles (K padded to 608 with zero weights)
#define NT   9       // 2 title n-tiles + 7 body n-tiles
#define MT   32      // 512/16 m-tiles
#define LSEQ (LT+LB) // 125

#define STAGE_TOTAL (KT * NT * 512)          // 87552 fragment elements
#define AFRAG_BYTES (MT * KT * 32 * 16 * 2)  // 622,592
#define TABLE_OFF   (AFRAG_BYTES + 1024)     // 1KB slack for A prefetch overrun

__device__ __forceinline__ float fast_tanh(float x) {
#if __has_builtin(__builtin_amdgcn_tanhf)
    return __builtin_amdgcn_tanhf(x);          // v_tanh_f32 (CDNA5 TRANS op)
#elif __has_builtin(__builtin_amdgcn_tanh_f32)
    return __builtin_amdgcn_tanh_f32(x);
#else
    return tanhf(x);
#endif
}

// ---------------------------------------------------------------------------
// Kernel 1: shuffle conv_w (f32, [H][C][W]) into f16 WMMA A-fragments.
// Layout: Afrag[((mt*KT + kt)*32 + lane)*16 + e], e=2v+j, with the ISA
// 16-bit A 16x32 mapping: k = (v&3)*2 + (v>>2)*16 + (lane>>4)*8 + j.
// K ordering chosen as k = w*200 + c (must match B staging). k>=600 -> 0.
// ---------------------------------------------------------------------------
__global__ __launch_bounds__(32) void prep_a_kernel(const float* __restrict__ cw,
                                                    _Float16* __restrict__ Afrag) {
    int blk  = blockIdx.x;            // 0 .. MT*KT-1
    int kt   = blk % KT;
    int mt   = blk / KT;
    int lane = threadIdx.x;           // 0..31
    int m    = mt * 16 + (lane & 15);
    int khalf = lane >> 4;
    _Float16* dst = Afrag + ((size_t)(mt * KT + kt) * 32 + lane) * 16;
    #pragma unroll
    for (int e = 0; e < 16; ++e) {
        int v = e >> 1, j = e & 1;
        int k = kt * 32 + (v & 3) * 2 + (v >> 2) * 16 + khalf * 8 + j;
        float val = 0.0f;
        if (k < KTOT) {
            int c = k % CC;
            int w = k / CC;
            val = cw[(m * CC + c) * WW + w];
        }
        dst[e] = (_Float16)val;
    }
}

// ---------------------------------------------------------------------------
// Kernel 1b: one-time im2col gather table. The staging mapping is identical
// for every sample (only the base pointer differs), so hoist the expensive
// idx -> source-offset decode out of the per-sample kernel. -1 = zero fill.
// Iteration order keeps 16 consecutive idx on 16 consecutive columns
// (consecutive l -> coalesced x reads in the staging loop).
// ---------------------------------------------------------------------------
__global__ __launch_bounds__(256) void prep_table_kernel(int* __restrict__ table) {
    int idx = blockIdx.x * 256 + threadIdx.x;
    if (idx >= STAGE_TOTAL) return;
    int col   = idx & 15;
    int khalf = (idx >> 4) & 1;
    int e     = (idx >> 5) & 15;
    int frag  = idx >> 9;             // kt*NT + nt
    int nt    = frag % NT;
    int kt    = frag / NT;
    int v = e >> 1, j = e & 1;
    int k = kt * 32 + (v & 3) * 2 + (v >> 2) * 16 + khalf * 8 + j;
    int off = -1;
    if (k < KTOT) {
        int c = k % CC;
        int w = k / CC;
        int segbase = (nt < 2) ? 0 : LT;
        int l       = ((nt < 2) ? nt * 16 : (nt - 2) * 16) + col;
        int pos     = segbase + l + w;
        if (pos > LSEQ - 1) pos = LSEQ - 1;   // padded cols: value unused
        off = c * LSEQ + pos;
    }
    table[idx] = off;
}

// ---------------------------------------------------------------------------
// Kernel 2: per-sample conv-as-GEMM with v_wmma_f32_16x16x32_f16.
// One workgroup (256 threads = 8 waves) per (b,n) sample.
// Each wave owns 4 m-tiles; per m-tile it holds NT=9 accumulators in
// registers and runs kt outer / nt inner: 1 A-fragment feeds 9 independent
// WMMAs per K-step (A loaded once per (mt,kt), not once per tile).
// ---------------------------------------------------------------------------
__global__ __launch_bounds__(256) void conv_gemm_kernel(
        const float* __restrict__ x,
        const float* __restrict__ pad_title,
        const float* __restrict__ pad_body,
        const float* __restrict__ conv_b,
        const _Float16* __restrict__ Afrag,
        const int* __restrict__ table,
        float* __restrict__ out_full)    // [B*N][512]
{
    __shared__ __align__(32) _Float16 Bfrag[KT * NT * 32 * 16]; // 175,104 B
    __shared__ float biasLds[HH];

    int s   = blockIdx.x;                 // sample index = b*22+n
    int tid = threadIdx.x;
    const float* xs = x + (size_t)s * CC * LSEQ;

    for (int i = tid; i < HH; i += 256) biasLds[i] = conv_b[i];

    // ---- Stage im2col B fragments into LDS via precomputed gather table.
    // dst bit-fields are disjoint: frag*512 | khalf*256 | col*16 | e.
    for (int idx = tid; idx < STAGE_TOTAL; idx += 256) {
        int off   = table[idx];
        float val = (off >= 0) ? xs[off] : 0.0f;
        int dst = (idx & ~511) | ((idx & 16) << 4) | ((idx & 15) << 4)
                | ((idx >> 5) & 15);
        Bfrag[dst] = (_Float16)val;
    }
    __syncthreads();

    int wave   = tid >> 5;
    int lane   = tid & 31;
    int col    = lane & 15;
    int rowgrp = lane >> 4;               // 0: M=e, 1: M=e+8
    float pt = pad_title[s];
    float pb = pad_body[s];
    float* op = out_full + (size_t)s * HH;

    #pragma unroll 1
    for (int i = 0; i < MT / 8; ++i) {
        int mt = wave * (MT / 8) + i;
        const _Float16* ap = Afrag + ((size_t)mt * KT * 32 + lane) * 16;
        const _Float16* bp = &Bfrag[(size_t)lane * 16];

        v8f acc[NT];
        #pragma unroll
        for (int nt = 0; nt < NT; ++nt) acc[nt] = (v8f){};

        // A prefetched one kt ahead; last prefetch reads 1KB past this
        // mt's fragments (within Afrag buffer slack, value never used).
        v16h a = *(const v16h*)(ap);
        #pragma unroll 1
        for (int kt = 0; kt < KT; ++kt) {
            v16h an = *(const v16h*)(ap + (size_t)(kt + 1) * 32 * 16);
            const _Float16* bk = bp + (size_t)kt * NT * 32 * 16;
            #pragma unroll
            for (int nt = 0; nt < NT; ++nt) {
                v16h bf = *(const v16h*)(bk + (size_t)nt * 32 * 16);
                acc[nt] = __builtin_amdgcn_wmma_f32_16x16x32_f16(
                              false, a, false, bf, (short)0, acc[nt], false, false);
            }
            a = an;
        }

        // ---- Epilogue: accumulate masked/scaled tanh into per-lane
        //      partials across all 9 n-tiles, then one butterfly per h. ----
        int hbase = mt * 16 + rowgrp * 8;
        float partial[8];
        #pragma unroll
        for (int e = 0; e < 8; ++e) partial[e] = 0.0f;

        #pragma unroll
        for (int nt = 0; nt < NT; ++nt) {
            int isTitle = (nt < 2);
            int l       = (isTitle ? nt * 16 : (nt - 2) * 16) + col;
            int lmax    = isTitle ? LOT : LOB;
            float validf  = (l < lmax) ? 1.0f : 0.0f;
            float segscal = 0.5f * (isTitle ? pt : pb) * validf;
            #pragma unroll
            for (int e = 0; e < 8; ++e)
                partial[e] += segscal * fast_tanh(acc[nt][e] + biasLds[hbase + e]);
        }

        #pragma unroll
        for (int e = 0; e < 8; ++e) {
            float z = partial[e];
            #pragma unroll
            for (int off = 8; off >= 1; off >>= 1)
                z += __shfl_xor(z, off, 16);
            if (col == 0) op[hbase + e] = z;   // unique (mt,rowgrp,e) per wave
        }
    }
}

// ---------------------------------------------------------------------------
// Kernel 3: cosine similarity matrix[b][q-1] = <out[b,q], out[b,0]> / norms.
// ---------------------------------------------------------------------------
__global__ __launch_bounds__(128) void cosine_kernel(const float* __restrict__ outbuf,
                                                     float* __restrict__ matrix) {
    int t = blockIdx.x;                  // 0 .. B*(N-1)-1
    int b = t / (NN - 1);
    int q = t % (NN - 1) + 1;
    const float* mv = outbuf + (size_t)(b * NN) * HH;
    const float* qv = outbuf + (size_t)(b * NN + q) * HH;
    float num = 0.f, qq = 0.f, mm = 0.f;
    for (int h = threadIdx.x; h < HH; h += 128) {
        float a = mv[h], c = qv[h];
        num += a * c; qq += c * c; mm += a * a;
    }
    #pragma unroll
    for (int off = 16; off >= 1; off >>= 1) {
        num += __shfl_xor(num, off);
        qq  += __shfl_xor(qq,  off);
        mm  += __shfl_xor(mm,  off);
    }
    __shared__ float rn[4], rq[4], rm[4];
    int wv = threadIdx.x >> 5;
    if ((threadIdx.x & 31) == 0) { rn[wv] = num; rq[wv] = qq; rm[wv] = mm; }
    __syncthreads();
    if (threadIdx.x == 0) {
        float N = 0.f, Qs = 0.f, M = 0.f;
        for (int i = 0; i < 4; ++i) { N += rn[i]; Qs += rq[i]; M += rm[i]; }
        float den = fmaxf(sqrtf(Qs) * sqrtf(M), 1e-8f);
        matrix[t] = N / den;
    }
}

// ---------------------------------------------------------------------------
extern "C" void kernel_launch(void* const* d_in, const int* in_sizes, int n_in,
                              void* d_out, int out_size, void* d_ws, size_t ws_size,
                              hipStream_t stream) {
    (void)in_sizes; (void)n_in; (void)out_size; (void)ws_size;
    const float* x      = (const float*)d_in[0];
    const float* pt     = (const float*)d_in[1];
    const float* pb     = (const float*)d_in[2];
    const float* conv_w = (const float*)d_in[3];
    const float* conv_b = (const float*)d_in[4];

    float* matrix   = (float*)d_out;                       // B*(N-1) = 2688
    float* out_full = (float*)d_out + BB * (NN - 1);       // B*N*H

    _Float16* Afrag = (_Float16*)d_ws;                     // 622,592 B
    int*      table = (int*)((char*)d_ws + TABLE_OFF);     // 350,208 B

    prep_a_kernel<<<MT * KT, 32, 0, stream>>>(conv_w, Afrag);
    prep_table_kernel<<<(STAGE_TOTAL + 255) / 256, 256, 0, stream>>>(table);
    conv_gemm_kernel<<<BB * NN, 256, 0, stream>>>(x, pt, pb, conv_b, Afrag, table,
                                                  out_full);
    cosine_kernel<<<BB * (NN - 1), 128, 0, stream>>>(out_full, matrix);
}
